// FocusedLSTMLayer_91079076479293
// MI455X (gfx1250) — compile-verified, matrix operand order checked by reading, not run on hardware
//
#include <hip/hip_runtime.h>
#include <hip/hip_bf16.h>

#define BATCH   64
#define SEQ     512
#define IN_DIM  512
#define UNITS   1024
#define ROWS    (BATCH * SEQ)        // 32768 flattened (b,s) rows

// Padded LDS row stride for the staged A tile: 1024 + 8 bf16 = 2064 bytes.
// Row r starts at bank (r*516)%64 = (r*4)%64 -> 16 rows cover all 64 banks,
// so the 16-lane ds_load_b128 fragment reads are conflict-free.
#define LDS_STRIDE 1032

#if defined(__has_builtin)
#if __has_builtin(__builtin_amdgcn_global_load_async_to_lds_b128)
#define HAVE_ASYNC_LDS 1
#endif
#endif

// ---- CDNA5 WMMA types -------------------------------------------------------
typedef __attribute__((ext_vector_type(16))) __bf16 bf16x16;
typedef __attribute__((ext_vector_type(8)))  float  f32x8;

// Exact pointee type the async-copy builtin wants (from hipcc diagnostic):
// int __attribute__((vector_size(16))) in AS1 (global) / AS3 (LDS).
typedef int v4i_b128 __attribute__((vector_size(16)));
typedef __attribute__((address_space(1))) v4i_b128* gas_b128_t;
typedef __attribute__((address_space(3))) v4i_b128* las_b128_t;

union TileCast {
    struct { uint4 lo, hi; } u;
    bf16x16 v;
};

// Load one wave's A/B fragment for v_wmma_f32_16x16x32_bf16 from a K-contiguous
// bf16 matrix (row-major, `stride` elements per row).  Per ISA 7.12.2:
//   lanes 0-15 : row = lane,    K = {0..7} U {16..23}
//   lanes 16-31: row = lane-16, K = {8..15} U {24..31}
// i.e. two 16-byte contiguous chunks per lane -> two global_load_b128.
__device__ __forceinline__ bf16x16 load_tile_bf16(const __hip_bfloat16* base,
                                                  int stride, int row, int kbase,
                                                  int lane) {
    const int r    = row + (lane & 15);
    const int koff = (lane >> 4) << 3;
    const __hip_bfloat16* p = base + (size_t)r * stride + kbase + koff;
    TileCast c;
    c.u.lo = *reinterpret_cast<const uint4*>(p);
    c.u.hi = *reinterpret_cast<const uint4*>(p + 16);
    return c.v;
}

// A-fragment from the LDS-staged tile (16 rows x 1024 cols, padded stride).
__device__ __forceinline__ bf16x16 load_tileA_lds(const __hip_bfloat16* sA,
                                                  int kbase, int lane) {
    const int r    = lane & 15;
    const int koff = (lane >> 4) << 3;
    const __hip_bfloat16* p = sA + r * LDS_STRIDE + kbase + koff;
    TileCast c;
    c.u.lo = *reinterpret_cast<const uint4*>(p);      // ds_load_b128
    c.u.hi = *reinterpret_cast<const uint4*>(p + 16); // ds_load_b128
    return c.v;
}

// Same fragment but sourced from fp32 memory, converted to bf16 on the fly.
__device__ __forceinline__ bf16x16 load_tile_f32(const float* base, int stride,
                                                 int row, int kbase, int lane) {
    const int r    = row + (lane & 15);
    const int koff = (lane >> 4) << 3;
    const float* p = base + (size_t)r * stride + kbase + koff;
    float4 a0 = *reinterpret_cast<const float4*>(p);
    float4 a1 = *reinterpret_cast<const float4*>(p + 4);
    float4 b0 = *reinterpret_cast<const float4*>(p + 16);
    float4 b1 = *reinterpret_cast<const float4*>(p + 20);
    bf16x16 v;
    v[0]  = (__bf16)a0.x; v[1]  = (__bf16)a0.y; v[2]  = (__bf16)a0.z; v[3]  = (__bf16)a0.w;
    v[4]  = (__bf16)a1.x; v[5]  = (__bf16)a1.y; v[6]  = (__bf16)a1.z; v[7]  = (__bf16)a1.w;
    v[8]  = (__bf16)b0.x; v[9]  = (__bf16)b0.y; v[10] = (__bf16)b0.z; v[11] = (__bf16)b0.w;
    v[12] = (__bf16)b1.x; v[13] = (__bf16)b1.y; v[14] = (__bf16)b1.z; v[15] = (__bf16)b1.w;
    return v;
}

__device__ __forceinline__ float sigmoidf_fast(float x) {
    return 1.0f / (1.0f + __expf(-x));
}
__device__ __forceinline__ float tanhf_fast(float x) {
    float xx = fminf(fmaxf(x, -15.0f), 15.0f);
    float e  = __expf(2.0f * xx);
    return (e - 1.0f) / (e + 1.0f);
}

// ---- workspace layout (bytes) ----------------------------------------------
//   [0,          4 MiB)  WrT  : recurrent kernel, transposed bf16 [2U][U]
//   [4 MiB,      5 MiB)  KT   : input kernel, transposed bf16 [U][D]
//   [5 MiB, +256 KiB  )  Ybuf : y ping-pong, bf16 2x[B][U]
//   then 8 bytes         Bar  : grid barrier {count, gen}
//   [6 MiB,     70 MiB)  Z    : z_all bf16 [B][S][U]
#define OFF_WRT  0
#define OFF_KT   (4u << 20)
#define OFF_Y    (5u << 20)
#define Y_BYTES  (2u * BATCH * UNITS * 2u)           // 262144
#define OFF_BAR  (OFF_Y + Y_BYTES)
#define OFF_Z    (6u << 20)

struct Bar { unsigned count; unsigned gen; };

// ---- kernel 0: zero y ping-pong buffers + barrier ---------------------------
__global__ void k_init(unsigned* p, int n_uints) {
    int i = blockIdx.x * blockDim.x + threadIdx.x;
    for (; i < n_uints; i += gridDim.x * blockDim.x) p[i] = 0u;
}

// ---- kernel 1: transpose-convert weights fp32 -> bf16 K-contiguous ----------
__global__ void k_convert(const float* __restrict__ kernel,
                          const float* __restrict__ rker,
                          __hip_bfloat16* __restrict__ kT,
                          __hip_bfloat16* __restrict__ wrT) {
    const size_t step = (size_t)gridDim.x * blockDim.x;
    size_t i0 = (size_t)blockIdx.x * blockDim.x + threadIdx.x;
    const size_t total1 = (size_t)UNITS * IN_DIM;          // kT[u][d] = kernel[d][u]
    for (size_t idx = i0; idx < total1; idx += step) {
        size_t u = idx / IN_DIM, d = idx % IN_DIM;
        kT[idx] = __float2bfloat16(kernel[d * UNITS + u]);
    }
    const size_t total2 = (size_t)2 * UNITS * UNITS;       // wrT[n][k] = rker[k][n]
    for (size_t idx = i0; idx < total2; idx += step) {
        size_t n = idx / UNITS, k = idx % UNITS;
        wrT[idx] = __float2bfloat16(rker[k * (2 * UNITS) + n]);
    }
}

// ---- kernel 2: z_all = tanh(X @ K + bz), bf16 WMMA, one 16x16 tile per wave -
__global__ void __launch_bounds__(256)
k_input_proj(const float* __restrict__ x, const __hip_bfloat16* __restrict__ kT,
             const float* __restrict__ bias, __hip_bfloat16* __restrict__ z) {
    const int lane = threadIdx.x & 31;
    const int wave = threadIdx.x >> 5;
    const int tile = blockIdx.x * (blockDim.x >> 5) + wave;   // 0..131071
    const int rt = tile >> 6;                                  // row tile 0..2047
    const int ct = tile & 63;                                  // col tile 0..63
    const int br = rt << 4;
    const int bu = ct << 4;

    f32x8 acc = {};
#pragma unroll 4
    for (int kb = 0; kb < IN_DIM; kb += 32) {
        bf16x16 a = load_tile_f32(x, IN_DIM, br, kb, lane);
        bf16x16 b = load_tile_bf16(kT, IN_DIM, bu, kb, lane);
        acc = __builtin_amdgcn_wmma_f32_16x16x32_bf16(false, a, false, b,
                                                      (short)0, acc, false, false);
    }

    // C/D layout: lane holds column n = lane&15, rows m = j + 8*(lane>>4)
    const int n  = lane & 15;
    const int mh = (lane >> 4) << 3;
    const float bz = bias[2 * UNITS + bu + n];
#pragma unroll
    for (int j = 0; j < 8; ++j) {
        float v = tanhf_fast(acc[j] + bz);
        z[(size_t)(br + mh + j) * UNITS + bu + n] = __float2bfloat16(v);
    }
}

// ---- grid barrier (generation counter) --------------------------------------
__device__ __forceinline__ void grid_sync(Bar* bar, unsigned nwg, unsigned target) {
    __builtin_amdgcn_fence(__ATOMIC_RELEASE, "agent");   // flush y_t stores
    __syncthreads();
    if (threadIdx.x == 0) {
        unsigned arrived = atomicAdd(&bar->count, 1u) + 1u;
        if (arrived == nwg) {
            bar->count = 0u;
            __threadfence();
            atomicAdd(&bar->gen, 1u);
        } else {
            while (__hip_atomic_load(&bar->gen, __ATOMIC_ACQUIRE,
                                     __HIP_MEMORY_SCOPE_AGENT) < target) {
                __builtin_amdgcn_s_sleep(1);
            }
        }
    }
    __syncthreads();
    __builtin_amdgcn_fence(__ATOMIC_ACQUIRE, "agent");   // invalidate before reading y_t
}

// ---- kernel 3: persistent recurrent scan ------------------------------------
// 32 blocks x 256 threads = 256 waves; wave w owns output tile (mt = w>>6,
// ut = w&63): 16 batch rows x 16 units.  All 8 waves of a block share mt, so
// the block stages its 16x1024 y_{t-1} slice into LDS once per step (async
// global->LDS DMA when available), then each wave computes BOTH the ri and ro
// 16x16 tiles (columns bu and UNITS+bu of r = y @ Wr) with 2x32 WMMAs per
// step, keeps c in registers for all 512 steps, and ping-pongs y through L2.
__global__ void __launch_bounds__(256)
k_scan(const __hip_bfloat16* __restrict__ z, const __hip_bfloat16* __restrict__ wrT,
       const float* __restrict__ bias, __hip_bfloat16* yb, float* __restrict__ y_out,
       Bar* bar) {
    __shared__ __align__(16) __hip_bfloat16 sA[16 * LDS_STRIDE];   // 33 KB

    const int lane = threadIdx.x & 31;
    const int wave = threadIdx.x >> 5;
    const int w    = blockIdx.x * (blockDim.x >> 5) + wave;   // 0..255
    const int mt = w >> 6;                                     // 0..3 (same for whole block)
    const int ut = w & 63;                                     // 0..63
    const int bm = mt << 4;
    const int bu = ut << 4;
    const int n  = lane & 15;
    const int mh = (lane >> 4) << 3;

    const float bi = bias[bu + n];
    const float bo = bias[UNITS + bu + n];

    float c[8];
#pragma unroll
    for (int j = 0; j < 8; ++j) c[j] = 0.0f;

    const unsigned nwg = gridDim.x;

    for (int t = 0; t < SEQ; ++t) {
        const __hip_bfloat16* yrd = yb + (size_t)(t & 1) * BATCH * UNITS;        // y_{t-1}
        __hip_bfloat16*       ywr = yb + (size_t)((t + 1) & 1) * BATCH * UNITS;  // y_t

        // ---- stage the block's 16x1024 A slice into LDS (2048 b128 chunks) --
#pragma unroll
        for (int chunk = 0; chunk < 8; ++chunk) {
            const int idx = threadIdx.x + chunk * 256;   // 0..2047
            const int row = idx >> 7;                    // 0..15
            const int col = (idx & 127) << 3;            // 0..1016 step 8
            const __hip_bfloat16* g = yrd + (size_t)(bm + row) * UNITS + col;
            __hip_bfloat16*       l = &sA[row * LDS_STRIDE + col];
#if defined(HAVE_ASYNC_LDS)
            __builtin_amdgcn_global_load_async_to_lds_b128(
                (gas_b128_t)g, (las_b128_t)l, 0, 0);
#else
            *reinterpret_cast<uint4*>(l) = *reinterpret_cast<const uint4*>(g);
#endif
        }
#if defined(HAVE_ASYNC_LDS)
        asm volatile("s_wait_asynccnt 0x0" ::: "memory");
#endif
        __syncthreads();

        // ---- r = y_{t-1} @ Wr for this wave's (bm, bu) tile pair ------------
        f32x8 ai = {}, ao = {};
#pragma unroll 2
        for (int kb = 0; kb < UNITS; kb += 32) {
            bf16x16 a  = load_tileA_lds(sA, kb, lane);
            bf16x16 b0 = load_tile_bf16(wrT, UNITS, bu, kb, lane);
            bf16x16 b1 = load_tile_bf16(wrT, UNITS, UNITS + bu, kb, lane);
            ai = __builtin_amdgcn_wmma_f32_16x16x32_bf16(false, a, false, b0,
                                                         (short)0, ai, false, false);
            ao = __builtin_amdgcn_wmma_f32_16x16x32_bf16(false, a, false, b1,
                                                         (short)0, ao, false, false);
        }

        // ---- gates + state update (c lives in registers) --------------------
#pragma unroll
        for (int j = 0; j < 8; ++j) {
            const int row = bm + mh + j;                     // batch index
            float ig = sigmoidf_fast(ai[j] + bi);
            float og = sigmoidf_fast(ao[j] + bo);
            float zt = __bfloat162float(z[((size_t)row * SEQ + t) * UNITS + bu + n]);
            c[j] += ig * zt;
            float yv = og * tanhf_fast(c[j]);
            ywr[(size_t)row * UNITS + bu + n] = __float2bfloat16(yv);
            if (t == SEQ - 1)
                y_out[(size_t)row * UNITS + bu + n] = yv;
        }

        grid_sync(bar, nwg, (unsigned)(t + 1));
    }
}

// ---- host-side launcher ------------------------------------------------------
extern "C" void kernel_launch(void* const* d_in, const int* in_sizes, int n_in,
                              void* d_out, int out_size, void* d_ws, size_t ws_size,
                              hipStream_t stream) {
    (void)in_sizes; (void)n_in; (void)out_size; (void)ws_size;
    const float* x    = (const float*)d_in[0];   // [64,512,512]
    const float* kern = (const float*)d_in[1];   // [512,1024]
    const float* rker = (const float*)d_in[2];   // [1024,2048]
    const float* bias = (const float*)d_in[3];   // [3072]
    float* y_out = (float*)d_out;                // [64,1024]

    char* ws = (char*)d_ws;
    __hip_bfloat16* wrT = (__hip_bfloat16*)(ws + OFF_WRT);
    __hip_bfloat16* kT  = (__hip_bfloat16*)(ws + OFF_KT);
    __hip_bfloat16* yb  = (__hip_bfloat16*)(ws + OFF_Y);
    Bar*            bar = (Bar*)(ws + OFF_BAR);
    __hip_bfloat16* z   = (__hip_bfloat16*)(ws + OFF_Z);

    // 0) zero y ping-pong + barrier (contiguous: Y_BYTES + 8 bytes)
    const int n_uints = (int)(Y_BYTES / 4) + 2;
    k_init<<<64, 256, 0, stream>>>((unsigned*)(ws + OFF_Y), n_uints);

    // 1) weights -> bf16, K-contiguous (transposed)
    k_convert<<<2048, 256, 0, stream>>>(kern, rker, kT, wrT);

    // 2) z_all = tanh(X @ K + bz): 2048x64 tiles, 8 waves/block
    k_input_proj<<<(ROWS / 16) * (UNITS / 16) / 8, 256, 0, stream>>>(x, kT, bias, z);

    // 3) persistent recurrent scan: 32 blocks x 8 waves = 256 tile-waves
    k_scan<<<32, 256, 0, stream>>>(z, wrT, bias, yb, y_out, bar);
}
